// SquaredL2Distance_27874337751124
// MI455X (gfx1250) — matrix operand order repaired and checked by reading, not compiled, not run on hardware
//
#include <hip/hip_runtime.h>

// Squared L2 distance matrix via f16 WMMA GEMM on gfx1250 (MI455X).
//   dists = qsq[:,None] + ssq[None,:] - 2 * Q @ S^T, clamped at 0.
// Pass 1: f32->f16 convert + exact f32 row norms (one wave32 per row).
// Pass 2: 128x128-tile WMMA GEMM, 3-stage GLOBAL_LOAD_ASYNC_TO_LDS_B128
//         pipeline (ASYNCcnt <= 4 drains only the older chunk), global
//         addresses bumped inside the asm (one loop-carried VGPR pair),
//         fully-unrolled k-loop, fused epilogue.
// Workspace: 2*8192*1024 halves (32MB) + 2*8192 f32 norms (64KB).

typedef __attribute__((ext_vector_type(16))) _Float16 v16h;
typedef __attribute__((ext_vector_type(8)))  _Float16 v8h;
typedef __attribute__((ext_vector_type(4)))  _Float16 v4h;
typedef __attribute__((ext_vector_type(8)))  float    v8f;

#define N_ROWS 8192
#define M_ROWS 8192
#define DDIM   1024

#define BM 128
#define BN 128
#define BK 32
#define LDT 40   // padded LDS row stride in halves (80B: 16B-aligned, conflict-free)
#define KSTEPS (DDIM / BK)
#define NBUF 3   // async pipeline depth

union Frag { v16h v; v8h h[2]; };

// ---------------- Pass 1: convert + squared norms ----------------
__global__ __launch_bounds__(256)
void SquaredL2_prep_kernel(const float* __restrict__ src,
                           _Float16* __restrict__ dsth,
                           float* __restrict__ sqnorm, int rows)
{
  int gid  = blockIdx.x * blockDim.x + threadIdx.x;
  int wave = gid >> 5;
  int lane = gid & 31;
  if (wave >= rows) return;

  const float* r = src  + (size_t)wave * DDIM;
  _Float16*    o = dsth + (size_t)wave * DDIM;

  float acc = 0.0f;
#pragma unroll
  for (int c = 0; c < DDIM / (32 * 4); ++c) {        // 8 float4 per lane
    int idx = (c * 32 + lane) * 4;
    float4 v = *(const float4*)(r + idx);
    acc += v.x * v.x + v.y * v.y + v.z * v.z + v.w * v.w;
    v4h h;
    h[0] = (_Float16)v.x; h[1] = (_Float16)v.y;
    h[2] = (_Float16)v.z; h[3] = (_Float16)v.w;
    *(v4h*)(o + idx) = h;                            // 8B store
  }
#pragma unroll
  for (int off = 16; off > 0; off >>= 1)
    acc += __shfl_xor(acc, off, 32);                 // wave32 reduction
  if (lane == 0) sqnorm[wave] = acc;
}

// Async global->LDS DMA of 32B (two b128 halves), then bump the global
// address by one k-chunk (64B) inside the asm so the address stays a single
// loop-carried VGPR pair (no hoisted per-chunk address blob).
// INST_OFFSET is added to both global and LDS addresses per the ISA, so the
// +16 split reuses one LDS/global base pair. Tracked by ASYNCcnt.
__device__ __forceinline__ void async_copy32_bump(unsigned lds_addr,
                                                  unsigned long long& gaddr) {
  asm volatile(
      "global_load_async_to_lds_b128 %1, %0, off\n\t"
      "global_load_async_to_lds_b128 %1, %0, off offset:16\n\t"
      "v_add_nc_u64 %0, 64, %0"
      : "+v"(gaddr)
      : "v"(lds_addr));
}

// Async loads complete in order per wave: waiting ASYNCcnt<=4 guarantees the
// chunk issued before the newest 4 ops has landed in LDS.
__device__ __forceinline__ void wait_async_le4() {
  asm volatile("s_wait_asynccnt 0x4" ::: "memory");
}
__device__ __forceinline__ void wait_async_le0() {
  asm volatile("s_wait_asynccnt 0x0" ::: "memory");
}

// ---------------- Pass 2: WMMA GEMM + fused L2 epilogue ----------------
__global__ __launch_bounds__(256)
void SquaredL2_wmma_kernel(const _Float16* __restrict__ Qh,
                           const _Float16* __restrict__ Sh,
                           const float* __restrict__ qsq,
                           const float* __restrict__ ssq,
                           float* __restrict__ out)
{
  __shared__ __align__(16) _Float16 lsA[NBUF][BM * LDT];   // 3 x 10KB
  __shared__ __align__(16) _Float16 lsB[NBUF][BN * LDT];   // 3 x 10KB

  const int t    = threadIdx.x;
  const int w    = t >> 5;        // wave id 0..7
  const int lane = t & 31;
  const int l16  = lane & 15;
  const int lh   = lane >> 4;     // half-wave select
  const int wm   = w & 3;         // 4 wave-rows  x 32 rows
  const int wn   = w >> 2;        // 2 wave-cols  x 64 cols

  const int rowBase = blockIdx.y * BM;   // query rows
  const int colBase = blockIdx.x * BN;   // source rows

  // Staging: thread t covers tile-row rr, 16-half segment seg (32B per matrix).
  const int rr  = t >> 1;
  const int seg = (t & 1) * 16;

  unsigned long long ga =
      (unsigned long long)(Qh + (size_t)(rowBase + rr) * DDIM + seg);
  unsigned long long gb =
      (unsigned long long)(Sh + (size_t)(colBase + rr) * DDIM + seg);

  // 32-bit LDS addresses: low 32 bits of the flat pointer are the DS offset.
  unsigned ldsA[NBUF], ldsB[NBUF];
#pragma unroll
  for (int i = 0; i < NBUF; ++i) {
    ldsA[i] = (unsigned)(size_t)&lsA[i][rr * LDT + seg];
    ldsB[i] = (unsigned)(size_t)&lsB[i][rr * LDT + seg];
  }

  // Prologue: DMA chunks 0 and 1 (addresses auto-bump inside the asm).
  async_copy32_bump(ldsA[0], ga);
  async_copy32_bump(ldsB[0], gb);
  async_copy32_bump(ldsA[1], ga);
  async_copy32_bump(ldsB[1], gb);
  wait_async_le4();                      // chunk 0 landed; chunk 1 in flight
  __syncthreads();

  v8f acc[2][4];
  {
    v8f z = {};
#pragma unroll
    for (int mb = 0; mb < 2; ++mb)
#pragma unroll
      for (int nb = 0; nb < 4; ++nb)
        acc[mb][nb] = z;
  }

  // Fully unrolled: buffer indices become compile-time, accumulators chain
  // in place inside the WMMA D=C register ranges.
#pragma unroll
  for (int kt = 0; kt < KSTEPS; ++kt) {
    const int buf = kt % NBUF;
    if (kt + 2 < KSTEPS) {               // DMA chunk kt+2, two iterations ahead
      const int nx = (kt + 2) % NBUF;
      async_copy32_bump(ldsA[nx], ga);
      async_copy32_bump(ldsB[nx], gb);
    }

    // A fragment (16x32 f16): lane l16 = row; lh selects K {0..7,16..23} vs {8..15,24..31}
    Frag afrag[2], bfrag[4];
#pragma unroll
    for (int mb = 0; mb < 2; ++mb) {
      const _Float16* p = &lsA[buf][(wm * 32 + mb * 16 + l16) * LDT];
      afrag[mb].h[0] = *(const v8h*)(p + lh * 8);
      afrag[mb].h[1] = *(const v8h*)(p + 16 + lh * 8);
    }
    // B fragment (32x16 f16): lane l16 = col (S row); lh selects K {0..7,16..23} vs {8..15,24..31}
#pragma unroll
    for (int nb = 0; nb < 4; ++nb) {
      const _Float16* p = &lsB[buf][(wn * 64 + nb * 16 + l16) * LDT];
      bfrag[nb].h[0] = *(const v8h*)(p + lh * 8);
      bfrag[nb].h[1] = *(const v8h*)(p + 16 + lh * 8);
    }

#pragma unroll
    for (int mb = 0; mb < 2; ++mb)
#pragma unroll
      for (int nb = 0; nb < 4; ++nb)
        acc[mb][nb] = __builtin_amdgcn_wmma_f32_16x16x32_f16(
            false, afrag[mb].v, false, bfrag[nb].v,
            (short)0, acc[mb][nb], false, false);

    if (kt + 1 < KSTEPS) {
      // Drain only the older in-flight chunk (kt+1, read next iteration);
      // the chunk just issued (kt+2) keeps flying.
      if (kt + 2 < KSTEPS) wait_async_le4();
      else                 wait_async_le0();
      __syncthreads();
    }
  }

  // Fused epilogue: qsq + ssq - 2*cross, clamp at 0.
  float sv[4];
#pragma unroll
  for (int nb = 0; nb < 4; ++nb)
    sv[nb] = ssq[colBase + wn * 64 + nb * 16 + l16];

#pragma unroll
  for (int mb = 0; mb < 2; ++mb) {
    const int rB = rowBase + wm * 32 + mb * 16 + 8 * lh;  // C: VGPR i -> M=i (+8 for hi half)
#pragma unroll
    for (int i = 0; i < 8; ++i) {
      const int row = rB + i;
      const float qv = qsq[row];
#pragma unroll
      for (int nb = 0; nb < 4; ++nb) {
        const int col = colBase + wn * 64 + nb * 16 + l16;
        float v = qv + sv[nb] - 2.0f * acc[mb][nb][i];
        v = fmaxf(v, 0.0f);
        out[(size_t)row * M_ROWS + col] = v;
      }
    }
  }
}

// ---------------- Host launch ----------------
extern "C" void kernel_launch(void* const* d_in, const int* in_sizes, int n_in,
                              void* d_out, int out_size, void* d_ws, size_t ws_size,
                              hipStream_t stream) {
  (void)in_sizes; (void)n_in; (void)out_size; (void)ws_size;

  const float* q = (const float*)d_in[0];   // [8192,1024] f32
  const float* s = (const float*)d_in[1];   // [8192,1024] f32
  float* out = (float*)d_out;               // [8192,8192] f32

  char* ws = (char*)d_ws;
  _Float16* Qh = (_Float16*)(ws);
  _Float16* Sh = (_Float16*)(ws + (size_t)N_ROWS * DDIM * sizeof(_Float16));
  float* qsq = (float*)(ws + (size_t)(N_ROWS + M_ROWS) * DDIM * sizeof(_Float16));
  float* ssq = qsq + N_ROWS;

  // one wave per row: 8192 waves = 1024 blocks of 256 threads
  dim3 pgrid(N_ROWS * 32 / 256);
  SquaredL2_prep_kernel<<<pgrid, 256, 0, stream>>>(q, Qh, qsq, N_ROWS);
  SquaredL2_prep_kernel<<<pgrid, 256, 0, stream>>>(s, Sh, ssq, M_ROWS);

  dim3 grid(M_ROWS / BN, N_ROWS / BM);      // 64 x 64 tiles
  SquaredL2_wmma_kernel<<<grid, 256, 0, stream>>>(Qh, Sh, qsq, ssq, out);
}
